// SCAM_49091476193887
// MI455X (gfx1250) — matrix operand order repaired and checked by reading, not compiled, not run on hardware
//
#include <hip/hip_runtime.h>
#include <math.h>

typedef __attribute__((ext_vector_type(16))) __bf16 v16bf;
typedef __attribute__((ext_vector_type(8)))  __bf16 v8bf;
typedef __attribute__((ext_vector_type(4)))  __bf16 v4bf;
typedef __attribute__((ext_vector_type(8)))  float  v8f;

#define EPSBN 1e-5f

// Single v_rcp_f32 (~1 ulp) instead of the IEEE divide expansion (~10 VALU ops)
__device__ __forceinline__ float fast_rcp(float x) { return __builtin_amdgcn_rcpf(x); }
__device__ __forceinline__ float sigmoid_f(float x) { return fast_rcp(1.f + __expf(-x)); }
__device__ __forceinline__ float silu_f(float x) { return x * sigmoid_f(x); }
__device__ __forceinline__ float gelu_f(float x) { return 0.5f * x * (1.f + erff(x * 0.70710678118654752f)); }

// ---------------------------------------------------------------------------
// Kernel 1: per-row (bn=16384) mean/max over d=128, written transposed so the
// gate MLP sees [b][p][n] contiguously.  grid=2048 x 256 (8 waves, 1 row/wave)
// ---------------------------------------------------------------------------
__global__ __launch_bounds__(256) void pool_stats_kernel(
    const float* __restrict__ x, float* __restrict__ avgT, float* __restrict__ maxT)
{
    const int wv   = threadIdx.x >> 5;
    const int lane = threadIdx.x & 31;
    const int row  = blockIdx.x * 8 + wv;          // 0..16383
    const int b    = row >> 8;
    const int nidx = row & 255;
    const float4* xr = (const float4*)(x + (size_t)row * 1536);
    #pragma unroll
    for (int p = 0; p < 12; ++p) {
        float4 v = xr[p * 32 + lane];
        float s = v.x + v.y + v.z + v.w;
        float m = fmaxf(fmaxf(v.x, v.y), fmaxf(v.z, v.w));
        #pragma unroll
        for (int off = 16; off >= 1; off >>= 1) {
            s += __shfl_xor(s, off);
            m = fmaxf(m, __shfl_xor(m, off));
        }
        if (lane == 0) {
            avgT[(b * 12 + p) * 256 + nidx] = s * (1.f / 128.f);
            maxT[(b * 12 + p) * 256 + nidx] = m;
        }
    }
}

// ---------------------------------------------------------------------------
// Kernel 2: per-batch gate MLPs (tiny, ~0.2 GFLOP total).  grid=64 x 256.
// pool[b][n][p] = sigmoid(gelu(avg@W1a^T)@W2a^T) + sigmoid(gelu(max@W1m^T)@W2m^T)
// ---------------------------------------------------------------------------
__global__ __launch_bounds__(256) void gates_kernel(
    const float* __restrict__ avgT, const float* __restrict__ maxT,
    const float* __restrict__ w1a, const float* __restrict__ w2a,
    const float* __restrict__ w1m, const float* __restrict__ w2m,
    float* __restrict__ pool)
{
    __shared__ float sA[12 * 256];
    __shared__ float sM[12 * 256];
    __shared__ float hA[12 * 128];
    __shared__ float hM[12 * 128];
    const int t = threadIdx.x;
    const int b = blockIdx.x;
    const float* aSrc = avgT + b * 3072;
    const float* mSrc = maxT + b * 3072;
    for (int i = t; i < 3072; i += 256) { sA[i] = aSrc[i]; sM[i] = mSrc[i]; }
    __syncthreads();
    for (int idx = t; idx < 1536; idx += 256) {       // [p][i] = [12][128]
        const int p = idx >> 7, i = idx & 127;
        const float* r1a = w1a + i * 256;
        const float* r1m = w1m + i * 256;
        const float* pa  = sA + p * 256;
        const float* pm  = sM + p * 256;
        float accA = 0.f, accM = 0.f;
        for (int nn = 0; nn < 256; ++nn) { accA += pa[nn] * r1a[nn]; accM += pm[nn] * r1m[nn]; }
        hA[idx] = gelu_f(accA); hM[idx] = gelu_f(accM);
    }
    __syncthreads();
    for (int idx = t; idx < 3072; idx += 256) {       // idx = n*12 + p
        const int nn = idx / 12, p = idx % 12;
        const float* r2a = w2a + nn * 128;
        const float* r2m = w2m + nn * 128;
        const float* ha  = hA + p * 128;
        const float* hm  = hM + p * 128;
        float accA = 0.f, accM = 0.f;
        for (int i = 0; i < 128; ++i) { accA += ha[i] * r2a[i]; accM += hm[i] * r2m[i]; }
        pool[b * 3072 + idx] = sigmoid_f(accA) + sigmoid_f(accM);
    }
}

// ---------------------------------------------------------------------------
// Kernel 3: main fused kernel.  128 threads = 4 waves; each wave does 4 rows.
// grid = 1024 -> 1024*4*4 = 16384 rows.
//   v = silu(BN(x @ wv^T)) via v_wmma_f32_16x16x32_bf16 (M=p padded to 16,
//   N=e in 8 tiles, K=d in 4 slices of 32), f32 accumulate + f32 epilogue.
// ---------------------------------------------------------------------------
__global__ __launch_bounds__(128) void scam_main_kernel(
    const float* __restrict__ x,  const float* __restrict__ wk,
    const float* __restrict__ bnk_g, const float* __restrict__ bnk_b,
    const float* __restrict__ bnk_rm, const float* __restrict__ bnk_rv,
    const float* __restrict__ wvw, const float* __restrict__ bnv_g,
    const float* __restrict__ bnv_b, const float* __restrict__ bnv_rm,
    const float* __restrict__ bnv_rv, const float* __restrict__ pool,
    float* __restrict__ out)
{
    __shared__ __align__(32) __bf16 s_wvbf[128 * 128];   // 32 KB, B-matrix source
    __shared__ float s_scale[128];
    __shared__ float s_shift[128];
    __shared__ float s_wk[12];
    __shared__ __align__(32) float s_area[4][1536];      // 6 KB per wave

    const int t    = threadIdx.x;
    const int wvid = t >> 5;
    const int lane = t & 31;

    // block-wide setup: bf16 copy of wv, BN(v) affine, wk
    for (int i = t; i < 16384; i += 128) s_wvbf[i] = (__bf16)wvw[i];  // wv[e][d] row-major
    if (t < 128) {
        float sc = bnv_g[t] * rsqrtf(bnv_rv[t] + EPSBN);
        s_scale[t] = sc;
        s_shift[t] = bnv_b[t] - bnv_rm[t] * sc;
    }
    if (t < 12) s_wk[t] = wk[t];
    __syncthreads();

    const float kaff_scale = bnk_g[0] * rsqrtf(bnk_rv[0] + EPSBN);
    const float kaff_shift = bnk_b[0] - bnk_rm[0] * kaff_scale;

    __bf16* xbf   = (__bf16*)s_area[wvid];   // [16][128] bf16 (4 KB), padded M
    float*  vtile = s_area[wvid];            // [12][128] f32 (6 KB), reused after

    const int m    = lane & 15;              // A-matrix row (p), lanes mirrored
    const int half = lane >> 4;

    for (int ri = 0; ri < 4; ++ri) {
        const int row = (blockIdx.x * 4 + wvid) * 4 + ri;   // 0..16383

        // ---- Phase A: load x row-tile, k_raw partials, bf16 copy into LDS
        float kraw[4] = {0.f, 0.f, 0.f, 0.f};              // d = 4*lane + j
        const float4* xr = (const float4*)(x + (size_t)row * 1536);
        #pragma unroll
        for (int p = 0; p < 12; ++p) {
            float4 v = xr[p * 32 + lane];
            float wkp = s_wk[p];
            kraw[0] += wkp * v.x; kraw[1] += wkp * v.y;
            kraw[2] += wkp * v.z; kraw[3] += wkp * v.w;
            v4bf pk = { (__bf16)v.x, (__bf16)v.y, (__bf16)v.z, (__bf16)v.w };
            *(v4bf*)(xbf + p * 128 + 4 * lane) = pk;
        }
        {   // zero-pad M rows 12..15
            v4bf z = { (__bf16)0.f, (__bf16)0.f, (__bf16)0.f, (__bf16)0.f };
            #pragma unroll
            for (int p = 12; p < 16; ++p) *(v4bf*)(xbf + p * 128 + 4 * lane) = z;
        }
        // prefetch next row's x tile (6 KB) so its loads overlap our WMMA work
        if (ri < 3) {
            const char* nx = (const char*)(x + (size_t)(row + 1) * 1536);
            __builtin_prefetch(nx + lane * 96, 0, 1);          // first 3 KB
            __builtin_prefetch(nx + 3072 + lane * 96, 0, 1);   // second 3 KB
        }
        // cross-lane LDS RAW: make all stores visible before A-fragment gather
        asm volatile("s_wait_dscnt 0" ::: "memory");

        // ---- A fragments (16x32 bf16 layout: element j -> K = half*16 + j (contiguous))
        v16bf afr[4];
        #pragma unroll
        for (int ks = 0; ks < 4; ++ks) {
            const int base = m * 128 + ks * 32 + half * 16;
            afr[ks] = *(const v16bf*)(xbf + base);
        }

        // ---- Phase B: WMMA over 8 N-tiles x 4 K-slices, BN+SiLU epilogue -> vtile
        const int n = lane & 15;            // B/D column
        #pragma unroll
        for (int nt = 0; nt < 8; ++nt) {
            v8f acc = {};
            #pragma unroll
            for (int ks = 0; ks < 4; ++ks) {
                // B 32x16 bf16 layout: element j -> K = half*16 + j (contiguous)
                const v16bf bfr = *(const v16bf*)(s_wvbf + (nt * 16 + n) * 128 + ks * 32 + half * 16);
                acc = __builtin_amdgcn_wmma_f32_16x16x32_bf16(
                    false, afr[ks], false, bfr, (short)0, acc, false, false);
            }
            const int e  = nt * 16 + n;
            const float sc = s_scale[e], sh = s_shift[e];
            // D layout: M = r + 8*half.  r=0..3 -> p in {0..3, 8..11}: always
            // valid, store unconditionally for all lanes.
            #pragma unroll
            for (int r = 0; r < 4; ++r) {
                float val = silu_f(acc[r] * sc + sh);
                vtile[(r + 8 * half) * 128 + e] = val;
            }
            // r=4..7 -> p in {4..7} (half=0, valid) or {12..15} (half=1, pad):
            // one uniform-per-halfwave branch instead of 4 predicated stores.
            if (half == 0) {
                #pragma unroll
                for (int r = 4; r < 8; ++r) {
                    float val = silu_f(acc[r] * sc + sh);
                    vtile[(r + 8 * half) * 128 + e] = val;
                }
            }
        }
        asm volatile("s_wait_dscnt 0" ::: "memory");

        // ---- Phase C: k = softmax(silu(affine(k_raw))) over d=128 (f32)
        float tt[4], mmax = -INFINITY;
        #pragma unroll
        for (int j = 0; j < 4; ++j) {
            tt[j] = silu_f(kraw[j] * kaff_scale + kaff_shift);
            mmax = fmaxf(mmax, tt[j]);
        }
        #pragma unroll
        for (int off = 16; off >= 1; off >>= 1) mmax = fmaxf(mmax, __shfl_xor(mmax, off));
        float k4[4], ssum = 0.f;
        #pragma unroll
        for (int j = 0; j < 4; ++j) { k4[j] = __expf(tt[j] - mmax); ssum += k4[j]; }
        #pragma unroll
        for (int off = 16; off >= 1; off >>= 1) ssum += __shfl_xor(ssum, off);
        const float inv = fast_rcp(ssum);
        #pragma unroll
        for (int j = 0; j < 4; ++j) k4[j] *= inv;     // k at d = 4*lane + j

        // ---- Phase D: y[p] = v.k ; y_pool[d] = pool.v ; out = y_pool (x) y
        float ypool[4] = {0.f, 0.f, 0.f, 0.f};
        float yv[12];
        const float* poolrow = pool + row * 12;
        #pragma unroll
        for (int p = 0; p < 12; ++p) {
            const float pl = poolrow[p];
            float yp = 0.f;
            #pragma unroll
            for (int j = 0; j < 4; ++j) {
                float vv = vtile[p * 128 + 4 * lane + j];
                yp += vv * k4[j];
                ypool[j] += pl * vv;
            }
            #pragma unroll
            for (int off = 16; off >= 1; off >>= 1) yp += __shfl_xor(yp, off);
            yv[p] = yp;
        }
        float4* orow = (float4*)(out + (size_t)row * 1536);
        #pragma unroll
        for (int p = 0; p < 12; ++p) {
            float4 o;
            o.x = yv[p] * ypool[0]; o.y = yv[p] * ypool[1];
            o.z = yv[p] * ypool[2]; o.w = yv[p] * ypool[3];
            orow[p * 32 + lane] = o;
        }
        // WAR guard: next iteration's xbf stores overwrite vtile just read
        asm volatile("s_wait_dscnt 0" ::: "memory");
    }
}

// ---------------------------------------------------------------------------
extern "C" void kernel_launch(void* const* d_in, const int* in_sizes, int n_in,
                              void* d_out, int out_size, void* d_ws, size_t ws_size,
                              hipStream_t stream) {
    const float* x      = (const float*)d_in[0];
    const float* wk     = (const float*)d_in[1];
    const float* bnk_g  = (const float*)d_in[2];
    const float* bnk_b  = (const float*)d_in[3];
    const float* bnk_rm = (const float*)d_in[4];
    const float* bnk_rv = (const float*)d_in[5];
    const float* wvw    = (const float*)d_in[6];
    const float* bnv_g  = (const float*)d_in[7];
    const float* bnv_b  = (const float*)d_in[8];
    const float* bnv_rm = (const float*)d_in[9];
    const float* bnv_rv = (const float*)d_in[10];
    const float* w_avg1 = (const float*)d_in[11];
    const float* w_avg2 = (const float*)d_in[12];
    const float* w_max1 = (const float*)d_in[13];
    const float* w_max2 = (const float*)d_in[14];

    float* ws   = (float*)d_ws;
    float* avgT = ws;                 // [64][12][256]
    float* maxT = ws + 196608;        // [64][12][256]
    float* pool = ws + 2 * 196608;    // [64][256][12]

    pool_stats_kernel<<<2048, 256, 0, stream>>>(x, avgT, maxT);
    gates_kernel<<<64, 256, 0, stream>>>(avgT, maxT, w_avg1, w_avg2, w_max1, w_max2, pool);
    scam_main_kernel<<<1024, 128, 0, stream>>>(
        x, wk, bnk_g, bnk_b, bnk_rm, bnk_rv,
        wvw, bnv_g, bnv_b, bnv_rm, bnv_rv, pool, (float*)d_out);
}
